// Efficient_Attention_1425929142540
// MI455X (gfx1250) — matrix operand order repaired and checked by reading, not compiled, not run on hardware
//
#include <hip/hip_runtime.h>

// ============================================================================
// Efficient (linear) attention for MI455X (gfx1250), wave32 + WMMA.
//
// B=4, C=128 (Cin=Ck=Cv), N=65536, 8 heads x dim 16. Memory-bound:
// ~36.5 GFLOP vs ~400MB minimum HBM traffic @ 23.3 TB/s. Plan:
//   - x read once (fp32), converted to f16 in LDS
//   - K and V never leave LDS; context accumulated via global f32 atomics
//   - Q is the only big intermediate (f16, 67MB round trip)
//   - attended + output projection fused (no agg buffer)
// All GEMMs: v_wmma_f32_16x16x32_f16. Fragments loaded as b128 LDS reads.
// Six independent WMMA chains per K-step to cover the 5-slot RAW hazard.
// ============================================================================

typedef _Float16 f16;
typedef __attribute__((ext_vector_type(16))) _Float16 v16h;
typedef __attribute__((ext_vector_type(8)))  float    v8f;
typedef __attribute__((ext_vector_type(4)))  unsigned int u4;

#define N_TOT 65536   // H*W
#define C     128     // Cin = Ck = Cv
#define NT    128     // spatial columns per workgroup
#define HEADS 8
#define HD    16      // head dim

#define WMMA_F16(A, B, Cacc) \
    __builtin_amdgcn_wmma_f32_16x16x32_f16(false, (A), false, (B), (short)0, (Cacc), false, false)

// ---- Fragment builders (CDNA5 wave32 VGPR layouts), b128 LDS loads ---------
// A (16x32 f16, MxK): lane L holds row m=L&15; halves [0..7] = K 8*(L>>4)+0..7,
// halves [8..15] = K 16+8*(L>>4)+0..7  -> two contiguous 16B chunks.
__device__ __forceinline__ v16h frag_a_rm(const f16* base, int ld,
                                          int row0, int col0, int lane) {
    const int m    = lane & 15;
    const int hsel = lane >> 4;
    const f16* r = base + (row0 + m) * ld + col0 + hsel * 8;
    union { v16h h; u4 q[2]; } u;
    u.q[0] = *(const u4*)(r);
    u.q[1] = *(const u4*)(r + 16);
    return u.h;
}

// B (32x16 f16, KxN): lane L holds col n=L&15; halves [0..7] = K 8*(L>>4)+0..7,
// halves [8..15] = K 16+8*(L>>4)+0..7 (strided gathers).
__device__ __forceinline__ v16h frag_b_rm(const f16* base, int ld,
                                          int k0, int n0, int lane) {
    const f16* r = base + n0 + (lane & 15);
    const int hsel = lane >> 4;
    v16h v;
#pragma unroll
    for (int s = 0; s < 16; ++s) {
        const int k = (s < 8) ? (hsel * 8 + s) : (16 + hsel * 8 + (s - 8));
        v[s] = r[(k0 + k) * ld];
    }
    return v;
}

// ---- Kernel 0: convert weights to f16 in ws, zero context accumulator ------
__global__ void ea_prep(const float* __restrict__ Wk, const float* __restrict__ Wq,
                        const float* __restrict__ Wv, const float* __restrict__ Wr,
                        f16* __restrict__ Wh, float* __restrict__ ctx) {
    const int idx = blockIdx.x * blockDim.x + threadIdx.x;
    if (idx < 4 * C * C) {
        const int which = idx >> 14;            // 16384 elems per matrix
        const int off   = idx & (C * C - 1);
        const float* src = (which == 0) ? Wk : (which == 1) ? Wq
                          : (which == 2) ? Wv : Wr;
        Wh[idx] = (f16)src[off];
    } else {
        const int c = idx - 4 * C * C;
        if (c < 4 * HEADS * HD * HD) ctx[c] = 0.f;
    }
}

// ---- Kernel 1: k/q/v projections + per-head context accumulation -----------
// LDS: sW[3*128*128] | sX[128*NT] | sK[128*NT] | sVT[NT*128] | sB[3*128 f32]
__global__ void ea_proj_ctx(const float* __restrict__ x,
                            const float* __restrict__ bk,
                            const float* __restrict__ bqv,
                            const float* __restrict__ bv,
                            const f16* __restrict__ Wh,     // [3][128*128]
                            float* __restrict__ ctx,        // [B][8][16][16]
                            f16* __restrict__ Qg) {         // [B][128][N] f16
    extern __shared__ unsigned char smem_u8[];
    f16*   sW  = (f16*)smem_u8;          // Wk|Wq|Wv
    f16*   sX  = sW + 3 * C * C;         // [c][col]
    f16*   sK  = sX + C * NT;            // [c][col]   (relu'd keys)
    f16*   sVT = sK + C * NT;            // [col][c]   (values, transposed)
    float* sB  = (float*)(sVT + NT * C); // bk|bq|bv

    const int tid  = threadIdx.x;
    const int lane = tid & 31;
    const int wid  = tid >> 5;
    const int b    = blockIdx.y;
    const int n0   = blockIdx.x * NT;

    // Stage weights (f16, 128-bit copies) and biases.
    {
        const uint4* src = (const uint4*)Wh;
        uint4*       dst = (uint4*)sW;
        for (int i = tid; i < (3 * C * C) / 8; i += blockDim.x) dst[i] = src[i];
        if (tid < C) {
            sB[tid]         = bk[tid];
            sB[C + tid]     = bqv[tid];
            sB[2 * C + tid] = bv[tid];
        }
    }
    // Stage x tile (fp32 -> f16), float4 loads for global_load_b128.
    {
        const float* xb = x + ((size_t)b * C) * N_TOT;
        for (int i4 = tid; i4 < (C * NT) / 4; i4 += blockDim.x) {
            const int c   = (i4 * 4) / NT;
            const int col = (i4 * 4) % NT;
            const float4 v = *(const float4*)(xb + (size_t)c * N_TOT + n0 + col);
            f16* d = sX + c * NT + col;
            d[0] = (f16)v.x; d[1] = (f16)v.y; d[2] = (f16)v.z; d[3] = (f16)v.w;
        }
        if (tid < C && n0 + NT < N_TOT)   // warm L2 for the neighboring tile
            __builtin_prefetch(xb + (size_t)tid * N_TOT + n0 + NT, 0, 0);
    }
    __syncthreads();

    // k/q/v GEMMs: each wave owns (1 M-tile x 2 N-tiles); 6 independent WMMA
    // chains per K-step -> RAW reuse distance >= 5 (no hazard nops needed).
    for (int p = wid; p < 32; p += 8) {
        const int mt = p >> 2;            // 0..7
        const int nt = (p & 3) * 2;       // 0,2,4,6
        v8f aK0 = {}, aK1 = {}, aQ0 = {}, aQ1 = {}, aV0 = {}, aV1 = {};
#pragma unroll
        for (int kk = 0; kk < 4; ++kk) {
            const v16h b0 = frag_b_rm(sX, NT, kk * 32, nt * 16, lane);
            const v16h b1 = frag_b_rm(sX, NT, kk * 32, (nt + 1) * 16, lane);
            const v16h ak = frag_a_rm(sW + 0 * C * C, C, mt * 16, kk * 32, lane);
            aK0 = WMMA_F16(ak, b0, aK0);
            aK1 = WMMA_F16(ak, b1, aK1);
            const v16h aq = frag_a_rm(sW + 1 * C * C, C, mt * 16, kk * 32, lane);
            aQ0 = WMMA_F16(aq, b0, aQ0);
            aQ1 = WMMA_F16(aq, b1, aQ1);
            const v16h av = frag_a_rm(sW + 2 * C * C, C, mt * 16, kk * 32, lane);
            aV0 = WMMA_F16(av, b0, aV0);
            aV1 = WMMA_F16(av, b1, aV1);
        }
        const int nloc = lane & 15;
        const int mhi  = (lane >> 4) * 8;
#pragma unroll
        for (int i = 0; i < 8; ++i) {
            const int m = mt * 16 + i + mhi;
            const float bkm = sB[m], bqm = sB[C + m], bvm = sB[2 * C + m];
            const int na = nt * 16 + nloc;
            const int nb = na + 16;
            sK[m * NT + na] = (f16)fmaxf(aK0[i] + bkm, 0.f);
            sK[m * NT + nb] = (f16)fmaxf(aK1[i] + bkm, 0.f);
            sVT[na * C + m] = (f16)(aV0[i] + bvm);
            sVT[nb * C + m] = (f16)(aV1[i] + bvm);
            const size_t qrow = ((size_t)(b * C + m) << 16) + n0;
            Qg[qrow + na] = (f16)fmaxf(aQ0[i] + bqm, 0.f);
            Qg[qrow + nb] = (f16)fmaxf(aQ1[i] + bqm, 0.f);
        }
    }
    __syncthreads();

    // Per-head context partial: ctx[h] += K_h[16xNT] * V_h^T[NTx16].
    {
        const int h = wid;                 // 8 waves <-> 8 heads
        v8f acc0 = {}, acc1 = {};          // split chunks across two chains
#pragma unroll
        for (int c = 0; c < NT / 64; ++c) {
            const v16h a0 = frag_a_rm(sK, NT, h * HD, (2 * c) * 32, lane);
            const v16h b0 = frag_b_rm(sVT, C, (2 * c) * 32, h * HD, lane);
            acc0 = WMMA_F16(a0, b0, acc0);
            const v16h a1 = frag_a_rm(sK, NT, h * HD, (2 * c + 1) * 32, lane);
            const v16h b1 = frag_b_rm(sVT, C, (2 * c + 1) * 32, h * HD, lane);
            acc1 = WMMA_F16(a1, b1, acc1);
        }
        float* cdst = ctx + ((size_t)(b * HEADS + h)) * HD * HD;
        const int nloc = lane & 15;
        const int mhi  = (lane >> 4) * 8;
#pragma unroll
        for (int i = 0; i < 8; ++i)
            atomicAdd(&cdst[(i + mhi) * HD + nloc], acc0[i] + acc1[i]);
    }
}

// ---- Kernel 2: attended = ctx^T @ q, fused with out = Wr @ attended + br ---
// LDS: sWr[128*128] | sQ[128*NT] | sAtt[128*NT] (f16) | sCtx[2048] | sBr[128]
__global__ void ea_att_out(const f16* __restrict__ Wrh,
                           const float* __restrict__ ctx,
                           const f16* __restrict__ Qg,
                           const float* __restrict__ br,
                           float* __restrict__ out) {
    extern __shared__ unsigned char smem_u8[];
    f16*   sWr  = (f16*)smem_u8;
    f16*   sQ   = sWr + C * C;                 // [c][col]
    f16*   sAtt = sQ + C * NT;                 // [vchan][col]
    float* sCtx = (float*)(sAtt + C * NT);     // [h][k][v] for this batch
    float* sBr  = sCtx + HEADS * HD * HD;

    const int tid  = threadIdx.x;
    const int lane = tid & 31;
    const int wid  = tid >> 5;
    const int b    = blockIdx.y;
    const int n0   = blockIdx.x * NT;

    {
        const uint4* src = (const uint4*)Wrh;
        uint4*       dst = (uint4*)sWr;
        for (int i = tid; i < (C * C) / 8; i += blockDim.x) dst[i] = src[i];
        if (tid < C) sBr[tid] = br[tid];
    }
    for (int u = tid; u < (C * NT) / 8; u += blockDim.x) {      // Q tile, b128 copies
        const int c   = (u * 8) / NT;
        const int col = (u * 8) % NT;
        *(uint4*)(sQ + c * NT + col) =
            *(const uint4*)(Qg + ((size_t)(b * C + c) << 16) + n0 + col);
    }
    {
        const float* cs = ctx + (size_t)b * HEADS * HD * HD;
        for (int i = tid; i < HEADS * HD * HD; i += blockDim.x) sCtx[i] = cs[i];
    }
    __syncthreads();

    // attended[v,n] = sum_k ctx[k,v] * q[k,n]; K=16 zero-padded to 32.
    for (int p = wid; p < 64; p += 8) {
        const int h  = p >> 3;
        const int nt = p & 7;
        v16h a;   // A = ctx^T, row m = vchan, col k = kchan (k>=16 -> 0)
        {
            const int m = lane & 15, hsel = lane >> 4;
#pragma unroll
            for (int j = 0; j < 8; ++j) {
                const int k = (j < 4) ? (hsel * 8 + 2 * j)
                                      : (16 + hsel * 8 + 2 * (j - 4));
                a[2 * j]     = (k     < HD) ? (f16)sCtx[(h * HD + k)     * HD + m] : (f16)0.f;
                a[2 * j + 1] = (k + 1 < HD) ? (f16)sCtx[(h * HD + k + 1) * HD + m] : (f16)0.f;
            }
        }
        v16h bqf; // B = q_h, col n = lane&15, table-order k; k >= 16 zero padding
        {
            const int hsel = lane >> 4;
#pragma unroll
            for (int j = 0; j < 16; ++j) {
                const int k = (j < 8) ? (hsel * 8 + j) : (16 + hsel * 8 + (j - 8));
                bqf[j] = (k < HD) ? sQ[(h * HD + k) * NT + nt * 16 + (lane & 15)] : (f16)0.f;
            }
        }
        v8f acc = {};
        acc = WMMA_F16(a, bqf, acc);
        const int nloc = lane & 15, mhi = (lane >> 4) * 8;
#pragma unroll
        for (int i = 0; i < 8; ++i)
            sAtt[(h * HD + i + mhi) * NT + nt * 16 + nloc] = (f16)acc[i];
    }
    __syncthreads();

    // out = Wr @ attended + br; 2 N-tiles per wave for 2 independent chains.
    for (int p = wid; p < 32; p += 8) {
        const int mt = p >> 2;
        const int nt = (p & 3) * 2;
        v8f a0 = {}, a1 = {};
#pragma unroll
        for (int kk = 0; kk < 4; ++kk) {
            const v16h aw = frag_a_rm(sWr, C, mt * 16, kk * 32, lane);
            const v16h b0 = frag_b_rm(sAtt, NT, kk * 32, nt * 16, lane);
            const v16h b1 = frag_b_rm(sAtt, NT, kk * 32, (nt + 1) * 16, lane);
            a0 = WMMA_F16(aw, b0, a0);
            a1 = WMMA_F16(aw, b1, a1);
        }
        const int nloc = lane & 15, mhi = (lane >> 4) * 8;
#pragma unroll
        for (int i = 0; i < 8; ++i) {
            const int m = mt * 16 + i + mhi;
            const size_t orow = ((size_t)(b * C + m) << 16) + n0;
            const float brm = sBr[m];
            out[orow + nt * 16 + nloc]      = a0[i] + brm;
            out[orow + nt * 16 + 16 + nloc] = a1[i] + brm;
        }
    }
}

// ---- Host launcher ---------------------------------------------------------
extern "C" void kernel_launch(void* const* d_in, const int* in_sizes, int n_in,
                              void* d_out, int out_size, void* d_ws, size_t ws_size,
                              hipStream_t stream) {
    (void)in_sizes; (void)n_in; (void)out_size; (void)ws_size;
    const float* x  = (const float*)d_in[0];
    const float* Wk = (const float*)d_in[1];
    const float* bk = (const float*)d_in[2];
    const float* Wq = (const float*)d_in[3];
    const float* bq = (const float*)d_in[4];
    const float* Wv = (const float*)d_in[5];
    const float* bv = (const float*)d_in[6];
    const float* Wr = (const float*)d_in[7];
    const float* br = (const float*)d_in[8];
    // d_in[9] = head_count (fixed at 8 per reference setup; baked into kernels)

    // Workspace layout: [Wk|Wq|Wv|Wr f16: 131072B][ctx f32: 32768B][Q f16: 64MB]
    unsigned char* ws = (unsigned char*)d_ws;
    f16*   Wh  = (f16*)ws;
    float* ctx = (float*)(ws + 131072);
    f16*   Qg  = (f16*)(ws + 131072 + 32768);
    float* out = (float*)d_out;

    ea_prep<<<288, 256, 0, stream>>>(Wk, Wq, Wv, Wr, Wh, ctx);   // 73728 threads exact

    dim3 grid(N_TOT / NT, 4);
    const size_t sh1 = (size_t)(3 * C * C + 3 * C * NT) * sizeof(f16)
                     + (size_t)(3 * C) * sizeof(float);                          // ~194 KB
    ea_proj_ctx<<<grid, 256, sh1, stream>>>(x, bk, bq, bv, Wh, ctx, Qg);

    const size_t sh3 = (size_t)(C * C + 2 * C * NT) * sizeof(f16)
                     + (size_t)(HEADS * HD * HD + C) * sizeof(float);            // ~105 KB
    ea_att_out<<<grid, 256, sh3, stream>>>(Wh + 3 * C * C, ctx, Qg, br, out);
}
